// CausalSelfAttention_6279242186881
// MI455X (gfx1250) — compile-verified, matrix operand order checked by reading, not run on hardware
//
#include <hip/hip_runtime.h>
#include <hip/hip_bf16.h>
#include <stdint.h>

typedef __attribute__((ext_vector_type(16))) __bf16 v16bf;
typedef __attribute__((ext_vector_type(8)))  __bf16 v8bf;
typedef __attribute__((ext_vector_type(4)))  __bf16 v4bf;
typedef __attribute__((ext_vector_type(8)))  float  v8f;
typedef __attribute__((ext_vector_type(4)))  int    v4i;

#define N_B   4
#define SEQ   2048
#define CD    1024
#define TCD   3072
#define NH    16
#define HD    64
#define BTOK  (N_B * SEQ)   // 8192 tokens

// ---- CDNA5 async global->LDS copy (ASYNCcnt) with safe fallback ------------
#if defined(__HIP_DEVICE_COMPILE__) && defined(__gfx1250__) && \
    __has_builtin(__builtin_amdgcn_global_load_async_to_lds_b128)
#define HAVE_ASYNC 1
#else
#define HAVE_ASYNC 0
#endif

__device__ __forceinline__ void cp16(__bf16* dstLds, const __bf16* srcGlobal) {
#if HAVE_ASYNC
  __builtin_amdgcn_global_load_async_to_lds_b128(
      (__attribute__((address_space(1))) v4i*)srcGlobal,
      (__attribute__((address_space(3))) v4i*)dstLds, 0, 0);
#else
  *(v8bf*)dstLds = *(const v8bf*)srcGlobal;
#endif
}

template <int N>
__device__ __forceinline__ void wait_async() {
#if HAVE_ASYNC
#if __has_builtin(__builtin_amdgcn_s_wait_asynccnt)
  __builtin_amdgcn_s_wait_asynccnt(N);
#else
  asm volatile("s_wait_asynccnt %0" :: "n"(N) : "memory");
#endif
#endif
}

// ---------------------------------------------------------------------------
__device__ __forceinline__ __bf16 f2bf(float f) {
  union { float f; uint32_t u; } c; c.f = f;
  uint32_t r = (c.u + 0x7FFFu + ((c.u >> 16) & 1u)) >> 16;
  union { uint16_t s; __bf16 b; } o; o.s = (uint16_t)r;
  return o.b;
}

__device__ __forceinline__ v16bf ldfrag(const __bf16* p0, const __bf16* p1) {
  union { v16bf v; v8bf h[2]; } u;
  u.h[0] = *(const v8bf*)p0;
  u.h[1] = *(const v8bf*)p1;
  return u.v;
}

__device__ __forceinline__ v8f wmma_bf16(v16bf a, v16bf b, v8f c) {
  return __builtin_amdgcn_wmma_f32_16x16x32_bf16(false, a, false, b, (short)0, c,
                                                 false, false);
}

// ---------------------------------------------------------------------------
// Prep 1: x fp32 -> bf16 (same [BT,C] layout).
// ---------------------------------------------------------------------------
__global__ __launch_bounds__(256) void conv_x_kernel(const float* __restrict__ X,
                                                     __bf16* __restrict__ Xb) {
  const size_t i = ((size_t)blockIdx.x * 256 + threadIdx.x) * 4;
  const float4 f = *(const float4*)(X + i);
  v4bf o; o[0]=f2bf(f.x); o[1]=f2bf(f.y); o[2]=f2bf(f.z); o[3]=f2bf(f.w);
  *(v4bf*)(Xb + i) = o;
}

// ---------------------------------------------------------------------------
// Prep 2: W [K=1024][N] fp32 -> Wt [N][1024] bf16 (transposed, K contiguous).
// ---------------------------------------------------------------------------
__global__ __launch_bounds__(256) void transpose_w_kernel(
    const float* __restrict__ W, __bf16* __restrict__ Wt, int N) {
  __shared__ __bf16 T[64][72];
  const int t = threadIdx.x;
  const int k0 = blockIdx.x * 64;
  const int n0 = blockIdx.y * 64;
  for (int p = 0; p < 4; ++p) {
    const int k = p*16 + (t >> 4), n4 = (t & 15) * 4;
    const float4 f = *(const float4*)(W + (size_t)(k0 + k) * N + n0 + n4);
    T[n4+0][k]=f2bf(f.x); T[n4+1][k]=f2bf(f.y);
    T[n4+2][k]=f2bf(f.z); T[n4+3][k]=f2bf(f.w);
  }
  __syncthreads();
  for (int p = 0; p < 2; ++p) {
    const int n = t >> 2, kc = (t & 3) * 8 + p * 32;
    *(v8bf*)(Wt + (size_t)(n0 + n) * CD + k0 + kc) = *(const v8bf*)&T[n][kc];
  }
}

// ---------------------------------------------------------------------------
// Kernel 1: qkv = Xb @ W_attn + b_attn -> Q/K [B,H,T,D], V [B,H,D,T] bf16.
// 128x128x32 tiles, async double-buffered LDS, 8 waves, 2x4 WMMA tiles/wave.
// ---------------------------------------------------------------------------
__global__ __launch_bounds__(256)
__attribute__((amdgpu_waves_per_eu(1)))
void qkv_gemm_kernel(
    const __bf16* __restrict__ Xb, const __bf16* __restrict__ Wt,
    const float* __restrict__ bias,
    __bf16* __restrict__ Qo, __bf16* __restrict__ Ko, __bf16* __restrict__ Vo)
{
  __shared__ __bf16 As[2][128][40];
  __shared__ __bf16 Bs[2][128][40];
  const int tid  = threadIdx.x;
  const int lane = tid & 31, wave = tid >> 5;
  const int g = lane >> 4, ln = lane & 15;
  const int rbase = (wave & 3) * 32, cbase = (wave >> 2) * 64;
  const int m0 = blockIdx.x * 128;
  const int n0 = blockIdx.y * 128;

  v8f acc[2][4] = {};

  auto issue = [&](int it, int bsel) {
    const int k0 = it * 32;
    for (int p = 0; p < 2; ++p) {
      const int q = tid + p * 256;             // 512 chunks of 16B per tile
      const int row = q >> 2, col = (q & 3) * 8;
      cp16(&As[bsel][row][col], Xb + (size_t)(m0 + row) * CD + k0 + col);
      cp16(&Bs[bsel][row][col], Wt + (size_t)(n0 + row) * CD + k0 + col);
    }
  };

  issue(0, 0);
  for (int it = 0; it < CD / 32; ++it) {
    const int cur = it & 1;
    if (it + 1 < CD / 32) { issue(it + 1, cur ^ 1); wait_async<4>(); }
    else                  { wait_async<0>(); }
    __syncthreads();
    v16bf af[2], bfm[4];
    for (int i = 0; i < 2; ++i) {
      const __bf16* r = &As[cur][rbase + i*16 + ln][0];
      af[i] = ldfrag(r + g*8, r + 16 + g*8);
    }
    for (int j = 0; j < 4; ++j) {
      const __bf16* r = &Bs[cur][cbase + j*16 + ln][0];
      bfm[j] = ldfrag(r + g*8, r + 16 + g*8);
    }
    for (int i = 0; i < 2; ++i)
      for (int j = 0; j < 4; ++j)
        acc[i][j] = wmma_bf16(af[i], bfm[j], acc[i][j]);
    __syncthreads();
  }

  // Epilogue: one base pointer per (i,j); walk r with a constant stride.
  for (int i = 0; i < 2; ++i) {
    const int trow0 = m0 + rbase + i*16 + 8*g;     // rows trow0 + r
    const int b = trow0 >> 11, t0 = trow0 & 2047;  // same batch for all 8 rows
    for (int j = 0; j < 4; ++j) {
      const int n = n0 + cbase + j*16 + ln;
      const float bn = bias[n];
      const int which = n >> 10;                   // uniform per block
      const int c = n & 1023, h = c >> 6, d = c & 63;
      const size_t bh = (size_t)(b * NH + h);
      if (which == 0) {
        __bf16* p = Qo + (bh * SEQ + t0) * HD + d;
        for (int r = 0; r < 8; ++r) p[r * HD] = f2bf((acc[i][j][r] + bn) * 0.125f);
      } else if (which == 1) {
        __bf16* p = Ko + (bh * SEQ + t0) * HD + d;
        for (int r = 0; r < 8; ++r) p[r * HD] = f2bf(acc[i][j][r] + bn);
      } else {
        __bf16* p = Vo + (bh * HD + d) * SEQ + t0;
        for (int r = 0; r < 8; ++r) p[r] = f2bf(acc[i][j][r] + bn);
      }
    }
  }
}

// ---------------------------------------------------------------------------
// Kernel 2: causal flash attention. Block = (q-tile of 64) x (b,h).
// 128 threads = 4 waves, 16 q rows per wave. K/V tiles async double-buffered.
// ---------------------------------------------------------------------------
__global__ __launch_bounds__(128)
__attribute__((amdgpu_waves_per_eu(1)))
void attn_kernel(
    const __bf16* __restrict__ Qo, const __bf16* __restrict__ Ko,
    const __bf16* __restrict__ Vo, __bf16* __restrict__ AO)
{
  __shared__ __bf16 KsB[2][64][72];    // K tile [k][d]
  __shared__ __bf16 VtB[2][64][72];    // V tile [d][k] (already transposed)
  __shared__ __bf16 Ps[4][16][72];     // per-wave P relayout scratch
  const int tid  = threadIdx.x;
  const int lane = tid & 31, w = tid >> 5;
  const int g = lane >> 4, ln = lane & 15;
  const int qtile = blockIdx.x;
  const int bh    = blockIdx.y;
  const int b = bh >> 4, h = bh & 15;
  const int qrow0 = qtile * 64 + w * 16;
  const size_t headBase = (size_t)bh * SEQ * HD;   // same extent for K and V

  v16bf qf[2];
  {
    const __bf16* qp = Qo + headBase + (size_t)(qrow0 + ln) * HD;
    qf[0] = ldfrag(qp + g*8,      qp + 16 + g*8);
    qf[1] = ldfrag(qp + 32 + g*8, qp + 48 + g*8);
  }

  v8f o[4] = {};
  float m_i[8], l_i[8];
  for (int r = 0; r < 8; ++r) { m_i[r] = -3.0e38f; l_i[r] = 0.0f; }

  auto issueKV = [&](int kv, int bsel) {
    const int kv0 = kv * 64;
    for (int p = 0; p < 4; ++p) {
      const int q = tid + p * 128;              // 512 chunks of 16B per tile
      const int row = q >> 3, col = (q & 7) * 8;
      cp16(&KsB[bsel][row][col], Ko + headBase + (size_t)(kv0 + row) * HD + col);
      cp16(&VtB[bsel][row][col], Vo + headBase + (size_t)row * SEQ + kv0 + col);
    }
  };

  issueKV(0, 0);
  for (int kv = 0; kv <= qtile; ++kv) {
    const int kv0 = kv * 64;
    const int cur = kv & 1;
    if (kv < qtile) { issueKV(kv + 1, cur ^ 1); wait_async<8>(); }
    else            { wait_async<0>(); }
    __syncthreads();

    // S = Q @ K^T  (Q pre-scaled by 1/sqrt(D))
    v8f s[4] = {};
    for (int j = 0; j < 4; ++j) {
      const __bf16* r = &KsB[cur][j*16 + ln][0];
      for (int kk = 0; kk < 2; ++kk) {
        v16bf kb = ldfrag(r + kk*32 + g*8, r + kk*32 + 16 + g*8);
        s[j] = wmma_bf16(qf[kk], kb, s[j]);
      }
    }
    if (kv == qtile) {   // mask only the diagonal tile
      for (int j = 0; j < 4; ++j) {
        const int kg = kv0 + j*16 + ln;
        for (int r = 0; r < 8; ++r)
          if (kg > qrow0 + r + 8*g) s[j][r] = -3.0e38f;
      }
    }

    // Online softmax (C-layout: row = r+8g, col = ln; width-16 reductions).
    for (int r = 0; r < 8; ++r) {
      float sm = fmaxf(fmaxf(s[0][r], s[1][r]), fmaxf(s[2][r], s[3][r]));
      for (int off = 8; off; off >>= 1) sm = fmaxf(sm, __shfl_xor(sm, off, 16));
      const float mn    = fmaxf(m_i[r], sm);
      const float alpha = __expf(m_i[r] - mn);
      m_i[r] = mn;
      float rs = 0.0f;
      for (int j = 0; j < 4; ++j) {
        const float p = __expf(s[j][r] - mn);
        s[j][r] = p; rs += p;
      }
      for (int off = 8; off; off >>= 1) rs += __shfl_xor(rs, off, 16);
      l_i[r] = l_i[r] * alpha + rs;
      for (int j = 0; j < 4; ++j) o[j][r] *= alpha;
      for (int j = 0; j < 4; ++j)
        Ps[w][r + 8*g][j*16 + ln] = f2bf(s[j][r]);
    }

    // O += P @ V
    v16bf pa[2];
    {
      const __bf16* r = &Ps[w][ln][0];
      pa[0] = ldfrag(r + g*8,      r + 16 + g*8);
      pa[1] = ldfrag(r + 32 + g*8, r + 48 + g*8);
    }
    for (int j = 0; j < 4; ++j) {
      const __bf16* r = &VtB[cur][j*16 + ln][0];
      for (int kk = 0; kk < 2; ++kk) {
        v16bf vb = ldfrag(r + kk*32 + g*8, r + kk*32 + 16 + g*8);
        o[j] = wmma_bf16(pa[kk], vb, o[j]);
      }
    }
    __syncthreads();
  }

  {
    const int t0 = qrow0 + 8*g;
    __bf16* p0 = AO + ((size_t)(b*SEQ + t0)) * CD + h*HD + ln;
    for (int r = 0; r < 8; ++r) {
      const float inv = 1.0f / l_i[r];
      __bf16* p = p0 + (size_t)r * CD;
      for (int j = 0; j < 4; ++j) p[j*16] = f2bf(o[j][r] * inv);
    }
  }
}

// ---------------------------------------------------------------------------
// Kernel 3: out = AO @ W_proj + b_proj (fp32). Same async tiling as kernel 1.
// ---------------------------------------------------------------------------
__global__ __launch_bounds__(256)
__attribute__((amdgpu_waves_per_eu(1)))
void proj_gemm_kernel(
    const __bf16* __restrict__ A, const __bf16* __restrict__ Wt,
    const float* __restrict__ bias, float* __restrict__ Out)
{
  __shared__ __bf16 As[2][128][40];
  __shared__ __bf16 Bs[2][128][40];
  const int tid  = threadIdx.x;
  const int lane = tid & 31, wave = tid >> 5;
  const int g = lane >> 4, ln = lane & 15;
  const int rbase = (wave & 3) * 32, cbase = (wave >> 2) * 64;
  const int m0 = blockIdx.x * 128;
  const int n0 = blockIdx.y * 128;

  v8f acc[2][4] = {};

  auto issue = [&](int it, int bsel) {
    const int k0 = it * 32;
    for (int p = 0; p < 2; ++p) {
      const int q = tid + p * 256;
      const int row = q >> 2, col = (q & 3) * 8;
      cp16(&As[bsel][row][col], A  + (size_t)(m0 + row) * CD + k0 + col);
      cp16(&Bs[bsel][row][col], Wt + (size_t)(n0 + row) * CD + k0 + col);
    }
  };

  issue(0, 0);
  for (int it = 0; it < CD / 32; ++it) {
    const int cur = it & 1;
    if (it + 1 < CD / 32) { issue(it + 1, cur ^ 1); wait_async<4>(); }
    else                  { wait_async<0>(); }
    __syncthreads();
    v16bf af[2], bfm[4];
    for (int i = 0; i < 2; ++i) {
      const __bf16* r = &As[cur][rbase + i*16 + ln][0];
      af[i] = ldfrag(r + g*8, r + 16 + g*8);
    }
    for (int j = 0; j < 4; ++j) {
      const __bf16* r = &Bs[cur][cbase + j*16 + ln][0];
      bfm[j] = ldfrag(r + g*8, r + 16 + g*8);
    }
    for (int i = 0; i < 2; ++i)
      for (int j = 0; j < 4; ++j)
        acc[i][j] = wmma_bf16(af[i], bfm[j], acc[i][j]);
    __syncthreads();
  }

  // Epilogue: contiguous column per (i,j), stride CD over rows.
  for (int i = 0; i < 2; ++i) {
    const int trow0 = m0 + rbase + i*16 + 8*g;
    for (int j = 0; j < 4; ++j) {
      const int n = n0 + cbase + j*16 + ln;
      const float bn = bias[n];
      float* p = Out + (size_t)trow0 * CD + n;
      for (int r = 0; r < 8; ++r) p[(size_t)r * CD] = acc[i][j][r] + bn;
    }
  }
}

// ---------------------------------------------------------------------------
extern "C" void kernel_launch(void* const* d_in, const int* in_sizes, int n_in,
                              void* d_out, int out_size, void* d_ws, size_t ws_size,
                              hipStream_t stream) {
  (void)in_sizes; (void)n_in; (void)out_size; (void)ws_size;
  const float* x      = (const float*)d_in[0];
  const float* W_attn = (const float*)d_in[1];
  const float* b_attn = (const float*)d_in[2];
  const float* W_proj = (const float*)d_in[3];
  const float* b_proj = (const float*)d_in[4];

  const size_t elems = (size_t)BTOK * CD;       // 8M
  __bf16* Qb  = (__bf16*)d_ws;                  // [B,H,T,D], pre-scaled
  __bf16* Kb  = Qb  + elems;                    // [B,H,T,D]
  __bf16* Vb  = Kb  + elems;                    // [B,H,D,T] (transposed)
  __bf16* AO  = Vb  + elems;                    // [B,T,C]
  __bf16* Xb  = AO  + elems;                    // [BT,C] bf16 copy of x
  __bf16* WtA = Xb  + elems;                    // [3C][C] W_attn^T bf16
  __bf16* WtP = WtA + (size_t)TCD * CD;         // [C][C]  W_proj^T bf16
  // total workspace: 88 MB

  conv_x_kernel<<<dim3((unsigned)(elems / (256 * 4))), 256, 0, stream>>>(x, Xb);
  transpose_w_kernel<<<dim3(CD/64, TCD/64), 256, 0, stream>>>(W_attn, WtA, TCD);
  transpose_w_kernel<<<dim3(CD/64, CD/64),  256, 0, stream>>>(W_proj, WtP, CD);

  qkv_gemm_kernel<<<dim3(BTOK/128, TCD/128), 256, 0, stream>>>(Xb, WtA, b_attn,
                                                               Qb, Kb, Vb);
  attn_kernel<<<dim3(SEQ/64, N_B*NH), 128, 0, stream>>>(Qb, Kb, Vb, AO);
  proj_gemm_kernel<<<dim3(BTOK/128, CD/128), 256, 0, stream>>>(AO, WtP, b_proj,
                                                               (float*)d_out);
}
